// FMFBlock_18605798327011
// MI455X (gfx1250) — matrix operand order repaired and checked by reading, not compiled, not run on hardware
//
#include <hip/hip_runtime.h>
#include <hip/hip_bf16.h>
#include <math.h>

// ---------------------------------------------------------------------------
// FMF block for MI455X (gfx1250): matmuls via v_wmma_f32_16x16x32_f16,
// weight panels via TDM tensor_load_to_lds, activation panels via
// global_load_async_to_lds_b128.
// ---------------------------------------------------------------------------

typedef __attribute__((ext_vector_type(16))) _Float16 v16h;
typedef __attribute__((ext_vector_type(8)))  float    v8f;
typedef __attribute__((ext_vector_type(4)))  unsigned int v4u;
typedef __attribute__((ext_vector_type(8)))  int      v8i;
typedef __attribute__((ext_vector_type(4)))  int      v4i;

#define NB   8
#define NC   96
#define NH   48
#define NW   48
#define NL   (NH * NW)          // 2304
#define NM   (NB * NL)          // 18432 total pixels
#define DST  16                 // D_STATE
#define DTR  6                  // DT_RANK

// ---- WMMA fragment helpers (wave32, 16x16x32 f16->f32) --------------------

// A operand: 16x32 (MxK) half, LDS layout [m][k] row-major, stride ld.
// Per-lane: row m = lane&15 ; K = {0..7,16..23} (lane<16) or {8..15,24..31}.
__device__ __forceinline__ v16h frag_a(const _Float16* t, int ld) {
  int lane = threadIdx.x & 31;
  int row  = lane & 15;
  int kb   = (lane & 16) ? 8 : 0;
  const _Float16* p = t + row * ld + kb;
  v16h r;
#pragma unroll
  for (int i = 0; i < 8; ++i) r[i] = p[i];
#pragma unroll
  for (int i = 0; i < 8; ++i) r[8 + i] = p[16 + i];
  return r;
}

// B operand: 32x16 (KxN) half, staged transposed in LDS as Bt[n][k], stride ld.
// Per-lane: col n = lane&15 ; K = {0..7,16..23} (lane<16) or {8..15,24..31}.
__device__ __forceinline__ v16h frag_b(const _Float16* t, int ld) {
  int lane = threadIdx.x & 31;
  int row  = lane & 15;
  int kb   = (lane & 16) ? 8 : 0;
  const _Float16* p = t + row * ld + kb;
  v16h r;
#pragma unroll
  for (int i = 0; i < 8; ++i) r[i] = p[i];
#pragma unroll
  for (int i = 0; i < 8; ++i) r[8 + i] = p[16 + i];
  return r;
}

__device__ __forceinline__ v8f wmma_f16(v16h a, v16h b, v8f c) {
  return __builtin_amdgcn_wmma_f32_16x16x32_f16(
      /*neg_a=*/false, a, /*neg_b=*/false, b,
      /*c_mod=*/(short)0, c, /*reuse_a=*/false, /*reuse_b=*/false);
}

__device__ __forceinline__ v8f zero8() {
  v8f z;
#pragma unroll
  for (int i = 0; i < 8; ++i) z[i] = 0.0f;
  return z;
}

__device__ __forceinline__ float silu(float v) { return v / (1.0f + expf(-v)); }

// ---- CDNA5 data movers -----------------------------------------------------

// Async global->LDS copy, 16 bytes per active lane (GV addressing, ASYNCcnt).
__device__ __forceinline__ void async_ld_b128(unsigned int lds_byte,
                                              const void* base,
                                              unsigned int byte_off) {
  unsigned long long addr = (unsigned long long)(uintptr_t)base + byte_off;
  asm volatile("global_load_async_to_lds_b128 %0, %1, off"
               :: "v"(lds_byte), "v"(addr)
               : "memory");
}
__device__ __forceinline__ void wait_async0() {
  asm volatile("s_wait_asynccnt 0" ::: "memory");
}

// TDM: linear DMA of `ndw` dwords from global to LDS (1-row 2-D descriptor).
// Issue from one wave; caller waits with s_wait_tensorcnt then barriers.
__device__ __forceinline__ void tdm_load_linear(unsigned int lds_byte,
                                                const void* gptr,
                                                unsigned int ndw) {
  unsigned long long ga = (unsigned long long)(uintptr_t)gptr;
  v4u g0;
  g0[0] = 1u;                                    // count=1 (user descriptor)
  g0[1] = lds_byte;                              // lds_addr
  g0[2] = (unsigned int)(ga & 0xFFFFFFFFu);      // global_addr[31:0]
  g0[3] = (unsigned int)((ga >> 32) & 0x1FFFFFFu) | (2u << 30);  // type=2
  v8i g1;
  g1[0] = (int)(2u << 16);                       // data_size = 2 -> 4B elems
  g1[1] = (int)((ndw & 0xFFFFu) << 16);          // tensor_dim0[15:0]
  g1[2] = (int)(((ndw >> 16) & 0xFFFFu) | (1u << 16));  // dim0 hi | tensor_dim1=1
  g1[3] = (int)((ndw & 0xFFFFu) << 16);          // tile_dim0
  g1[4] = 1;                                     // tile_dim1 = 1
  g1[5] = (int)ndw;                              // tensor_dim0_stride lo
  g1[6] = 0;
  g1[7] = 0;
  v4i gz;
  gz[0] = 0; gz[1] = 0; gz[2] = 0; gz[3] = 0;
  v8i gz8;
#pragma unroll
  for (int i = 0; i < 8; ++i) gz8[i] = 0;
  __builtin_amdgcn_tensor_load_to_lds(g0, g1, gz, gz, gz8, 0);
}

__device__ __forceinline__ unsigned int lds_off(const void* p) {
  return (unsigned int)(uintptr_t)p;  // low 32 bits = LDS byte offset
}

// ---- prep kernels ----------------------------------------------------------

__global__ void prep_dct_kernel(_Float16* Dpad, float* mask) {
  int idx = blockIdx.x * blockDim.x + threadIdx.x;
  if (idx < 48 * 64) {
    int i = idx >> 6, j = idx & 63;
    float v = 0.0f;
    if (j < 48) {
      float s = (i == 0) ? sqrtf(1.0f / 48.0f) : sqrtf(2.0f / 48.0f);
      v = cosf(3.14159265358979323846f * (2.0f * j + 1.0f) * i / 96.0f) * s;
    }
    Dpad[idx] = (_Float16)v;
  } else if (idx < 48 * 64 + 48 * 48) {
    int k = idx - 48 * 64;
    int h = k / 48, w = k % 48;
    float uu = (h / 48.0f) * (h / 48.0f);
    float vv = (w / 48.0f) * (w / 48.0f);
    mask[k] = expf(-(uu + vv) / (2.0f * 0.15f * 0.15f));
  }
}

__global__ void conv_w_prep_kernel(const float* w, _Float16* wt, int ks) {
  int idx = blockIdx.x * blockDim.x + threadIdx.x;
  int total = ks * ks * NC * NC;
  if (idx >= total) return;
  int tap = idx / (NC * NC);
  int rem = idx % (NC * NC);
  int n = rem / NC, c = rem % NC;
  int kh = tap / ks, kw = tap % ks;
  wt[idx] = (_Float16)w[(((size_t)n * NC + c) * ks + kh) * ks + kw];
}

__global__ void mat_t_prep_kernel(const float* src, _Float16* dst,
                                  int K, int N, int Npad) {
  int idx = blockIdx.x * blockDim.x + threadIdx.x;
  if (idx >= Npad * K) return;
  int n = idx / K, k = idx % K;
  dst[idx] = (n < N) ? (_Float16)src[(size_t)k * N + n] : (_Float16)0.0f;
}

__global__ void mat_c_prep_kernel(const float* src, _Float16* dst, int total) {
  int idx = blockIdx.x * blockDim.x + threadIdx.x;
  if (idx < total) dst[idx] = (_Float16)src[idx];
}

// ---- DCT: fre = D @ x @ D^T, masked l/h split, write [b][l][c] half --------

__global__ __launch_bounds__(256) void dct_kernel(
    const float* __restrict__ x, const _Float16* __restrict__ Dpad,
    const float* __restrict__ mask,
    _Float16* __restrict__ lfre, _Float16* __restrict__ hfre) {
  __shared__ _Float16 sD[48 * 64];   // D[m][k] (also D[n][k] for pass 2)
  __shared__ _Float16 sXt[48 * 64];  // Xt[n][k] = x[k][n]
  __shared__ _Float16 sT[48 * 64];   // T[m][k]
  int bc = blockIdx.x;
  int b = bc / NC, c = bc % NC;
  const float* xs = x + (size_t)bc * NL;
  int tid = threadIdx.x;
  if (tid < 32) {  // wave 0: DMA the DCT matrix into LDS via TDM
    tdm_load_linear(lds_off(sD), Dpad, 48 * 64 * 2 / 4);
  }
  for (int i = tid; i < 48 * 64; i += 256) {
    sT[i] = (_Float16)0.0f;
    int n = i >> 6, k = i & 63;
    sXt[i] = (k < 48) ? (_Float16)xs[k * 48 + n] : (_Float16)0.0f;
  }
  if (tid < 32) __builtin_amdgcn_s_wait_tensorcnt(0);
  __syncthreads();
  int wave = tid >> 5;
  int lane = tid & 31;
  int mo = (lane & 16) ? 8 : 0;
  // pass 1: T = D @ X   (A = sD, Bt = sXt)
  for (int tile = wave; tile < 9; tile += 8) {
    int mt = tile / 3, nt = tile % 3;
    v8f acc = zero8();
#pragma unroll
    for (int kc = 0; kc < 2; ++kc) {
      v16h a  = frag_a(sD  + mt * 16 * 64 + kc * 32, 64);
      v16h bf = frag_b(sXt + nt * 16 * 64 + kc * 32, 64);
      acc = wmma_f16(a, bf, acc);
    }
    int n = nt * 16 + (lane & 15);
#pragma unroll
    for (int r = 0; r < 8; ++r)
      sT[(mt * 16 + r + mo) * 64 + n] = (_Float16)acc[r];
  }
  __syncthreads();
  // pass 2: F = T @ D^T  (A = sT, Bt[n][k] = D[n][k] = sD)
  for (int tile = wave; tile < 9; tile += 8) {
    int mt = tile / 3, nt = tile % 3;
    v8f acc = zero8();
#pragma unroll
    for (int kc = 0; kc < 2; ++kc) {
      v16h a  = frag_a(sT + mt * 16 * 64 + kc * 32, 64);
      v16h bf = frag_b(sD + nt * 16 * 64 + kc * 32, 64);
      acc = wmma_f16(a, bf, acc);
    }
    int n = nt * 16 + (lane & 15);
#pragma unroll
    for (int r = 0; r < 8; ++r) {
      int m = mt * 16 + r + mo;
      float f  = acc[r];
      float mk = mask[m * 48 + n];
      size_t o = ((size_t)b * NL + m * 48 + n) * NC + c;
      lfre[o] = (_Float16)(f * mk);
      hfre[o] = (_Float16)(f * (1.0f - mk));
    }
  }
}

// ---- conv as shifted implicit GEMM + silu, output [b][l][c] half -----------

template <int KS, int PAD>
__global__ __launch_bounds__(256) void conv_silu_kernel(
    const float* __restrict__ x, const _Float16* __restrict__ Wt,
    const float* __restrict__ bias, _Float16* __restrict__ out) {
  __shared__ _Float16 sA[64 * 32];
  __shared__ _Float16 sB[96 * 96];   // one full tap: Bt[n=cout][k=cin]
  int tid = threadIdx.x, wave = tid >> 5;
  int pixbase = blockIdx.x * 64;
  int b = pixbase / NL, lbase = pixbase % NL;
  int mt = wave & 3;
  int ng = (wave >> 2) * 3;
  v8f acc[3];
  acc[0] = zero8(); acc[1] = zero8(); acc[2] = zero8();
  for (int tap = 0; tap < KS * KS; ++tap) {
    int dh = tap / KS - PAD, dw = tap % KS - PAD;
    __syncthreads();  // previous tap's sB reads complete
    if (tid < 32) {   // TDM: whole tap weight panel (18KB) -> LDS
      tdm_load_linear(lds_off(sB), Wt + (size_t)tap * NC * NC,
                      NC * NC * 2 / 4);
      __builtin_amdgcn_s_wait_tensorcnt(0);
    }
    for (int kc = 0; kc < 3; ++kc) {
      __syncthreads();
      for (int i = tid; i < 64 * 32; i += 256) {
        int p = i >> 5, cc = i & 31;
        int l = lbase + p;
        int hh = l / 48 + dh, ww = l % 48 + dw;
        float v = 0.0f;
        if (hh >= 0 && hh < 48 && ww >= 0 && ww < 48)
          v = x[(((size_t)b * NC + kc * 32 + cc) * 48 + hh) * 48 + ww];
        sA[i] = (_Float16)v;
      }
      __syncthreads();
      v16h a = frag_a(sA + mt * 16 * 32, 32);
#pragma unroll
      for (int j = 0; j < 3; ++j) {
        v16h bf = frag_b(sB + (ng + j) * 16 * NC + kc * 32, NC);
        acc[j] = wmma_f16(a, bf, acc[j]);
      }
    }
  }
  int lane = tid & 31;
  int mo = (lane & 16) ? 8 : 0;
#pragma unroll
  for (int j = 0; j < 3; ++j) {
    int n = (ng + j) * 16 + (lane & 15);
    float bi = bias[n];
#pragma unroll
    for (int r = 0; r < 8; ++r) {
      int m = mt * 16 + mo + r;
      float v = acc[j][r] + bi;
      out[(size_t)(pixbase + m) * NC + n] = (_Float16)silu(v);
    }
  }
}

// ---- generic M64xN96 GEMM, K = 96 or 192, runtime epilogue mode ------------
// mode 0: silu -> outf (f32, optional) and outh (half, optional)
// mode 1: raw  -> outh
// mode 2: +bias, silu -> d_out f32 in [b][c][l] layout
// mode 3: g = sigmoid(+bias); outh = g*auxL + (1-g)*auxH
__global__ __launch_bounds__(256) void gemm96_kernel(
    const _Float16* __restrict__ A, const _Float16* __restrict__ A2,
    const _Float16* __restrict__ Bt, const float* __restrict__ bias,
    float* __restrict__ outf, _Float16* __restrict__ outh,
    const _Float16* __restrict__ auxL, const _Float16* __restrict__ auxH,
    int mode) {
  __shared__ _Float16 sA[64 * 32];
  __shared__ _Float16 sB[96 * 192];  // whole Bt panel (18 or 36 KB)
  int tid = threadIdx.x, wave = tid >> 5;
  int pixbase = blockIdx.x * 64;
  int mt = wave & 3;
  int ng = (wave >> 2) * 3;
  int nchunk = A2 ? 2 : 1;
  int ldb = 96 * nchunk;
  if (tid < 32) {  // TDM: full weight panel once per block
    tdm_load_linear(lds_off(sB), Bt, (unsigned int)(96 * ldb * 2 / 4));
    __builtin_amdgcn_s_wait_tensorcnt(0);
  }
  v8f acc[3];
  acc[0] = zero8(); acc[1] = zero8(); acc[2] = zero8();
  int p = tid >> 2, q = tid & 3;  // 256 threads x 16B = 4KB A panel
  for (int ch = 0; ch < nchunk; ++ch) {
    const _Float16* Ac = ch ? A2 : A;
    for (int kc = 0; kc < 3; ++kc) {
      __syncthreads();
      // async copy A panel: rows pixbase..+63, halves kc*32..+31
      async_ld_b128(lds_off(sA) + (unsigned int)(p * 64 + q * 16), Ac,
                    (unsigned int)(((pixbase + p) * NC + kc * 32 + q * 8) * 2));
      wait_async0();
      __syncthreads();
      v16h a = frag_a(sA + mt * 16 * 32, 32);
#pragma unroll
      for (int j = 0; j < 3; ++j) {
        v16h bf = frag_b(sB + (ng + j) * 16 * ldb + ch * 96 + kc * 32, ldb);
        acc[j] = wmma_f16(a, bf, acc[j]);
      }
    }
  }
  int lane = tid & 31;
  int mo = (lane & 16) ? 8 : 0;
  int b = pixbase / NL, lb = pixbase % NL;
#pragma unroll
  for (int j = 0; j < 3; ++j) {
    int n = (ng + j) * 16 + (lane & 15);
#pragma unroll
    for (int r = 0; r < 8; ++r) {
      int m = mt * 16 + mo + r;
      size_t pix = pixbase + m;
      float v = acc[j][r];
      if (mode == 0) {
        v = silu(v);
        if (outf) outf[pix * NC + n] = v;
        if (outh) outh[pix * NC + n] = (_Float16)v;
      } else if (mode == 1) {
        outh[pix * NC + n] = (_Float16)v;
      } else if (mode == 2) {
        v = silu(v + bias[n]);
        outf[((size_t)b * NC + n) * NL + lb + m] = v;
      } else {
        float g  = 1.0f / (1.0f + expf(-(v + bias[n])));
        float lv = (float)auxL[pix * NC + n];
        float hv = (float)auxH[pix * NC + n];
        outh[pix * NC + n] = (_Float16)(g * lv + (1.0f - g) * hv);
      }
    }
  }
}

// ---- small-N GEMM: M64 x N32 (Bt padded to 32 rows), K = 96 ----------------
__global__ __launch_bounds__(256) void gemm32_kernel(
    const _Float16* __restrict__ A, const _Float16* __restrict__ Bt,
    float* __restrict__ outf, int ncols) {
  __shared__ _Float16 sA[64 * 32];
  __shared__ _Float16 sB[32 * 96];
  int tid = threadIdx.x, wave = tid >> 5;
  int pixbase = blockIdx.x * 64;
  int mt = wave & 3, nt = wave >> 2;  // 4 x 2 tiles
  if (tid < 32) {
    tdm_load_linear(lds_off(sB), Bt, 32 * 96 * 2 / 4);
    __builtin_amdgcn_s_wait_tensorcnt(0);
  }
  v8f acc = zero8();
  int p = tid >> 2, q = tid & 3;
  for (int kc = 0; kc < 3; ++kc) {
    __syncthreads();
    async_ld_b128(lds_off(sA) + (unsigned int)(p * 64 + q * 16), A,
                  (unsigned int)(((pixbase + p) * NC + kc * 32 + q * 8) * 2));
    wait_async0();
    __syncthreads();
    v16h a  = frag_a(sA + mt * 16 * 32, 32);
    v16h bf = frag_b(sB + nt * 16 * 96 + kc * 32, 96);
    acc = wmma_f16(a, bf, acc);
  }
  int lane = tid & 31;
  int mo = (lane & 16) ? 8 : 0;
  int n = nt * 16 + (lane & 15);
  if (n < ncols) {
#pragma unroll
    for (int r = 0; r < 8; ++r) {
      int m = mt * 16 + mo + r;
      outf[(size_t)(pixbase + m) * ncols + n] = acc[r];
    }
  }
}

// ---- dt = softplus(dtB[:, :6] @ Wdt + bdt) ---------------------------------
__global__ void dt_kernel(const float* __restrict__ dtB,
                          const float* __restrict__ Wdt,
                          const float* __restrict__ bdt,
                          float* __restrict__ dt) {
  int idx = blockIdx.x * blockDim.x + threadIdx.x;
  if (idx >= NM * NC) return;
  int row = idx / NC, d = idx % NC;
  float s = bdt[d];
  const float* r = dtB + (size_t)row * (DTR + DST);
#pragma unroll
  for (int j = 0; j < DTR; ++j) s += r[j] * Wdt[j * NC + d];
  dt[idx] = (s > 20.0f) ? s : log1pf(expf(s));
}

// ---- selective scan: one thread per (b, d); state h[16] in registers -------
__global__ __launch_bounds__(96) void scan_kernel(
    const float* __restrict__ xf, const float* __restrict__ dt,
    const float* __restrict__ dtB, const float* __restrict__ Cp,
    const float* __restrict__ A_log, const float* __restrict__ Dp,
    const float* __restrict__ zsilu, _Float16* __restrict__ yh) {
  int b = blockIdx.x;
  int d = threadIdx.x;  // 96 threads
  float A[DST], h[DST];
#pragma unroll
  for (int n = 0; n < DST; ++n) {
    A[n] = -expf(A_log[d * DST + n]);
    h[n] = 0.0f;
  }
  float Dv = Dp[d];
  size_t base = (size_t)b * NL;
  for (int l = 0; l < NL; ++l) {
    size_t row = base + l;
    float dtv = dt[row * NC + d];
    float xv  = xf[row * NC + d];
    float dx  = dtv * xv;
    const float* Bp = dtB + row * (DTR + DST) + DTR;
    const float* Cr = Cp + row * DST;
    float y = 0.0f;
#pragma unroll
    for (int n = 0; n < DST; ++n) {
      h[n] = expf(dtv * A[n]) * h[n] + dx * Bp[n];
      y += h[n] * Cr[n];
    }
    y = (y + xv * Dv) * zsilu[row * NC + d];
    yh[row * NC + d] = (_Float16)y;
  }
}

// ---------------------------------------------------------------------------

extern "C" void kernel_launch(void* const* d_in, const int* in_sizes, int n_in,
                              void* d_out, int out_size, void* d_ws,
                              size_t ws_size, hipStream_t stream) {
  (void)in_sizes; (void)n_in; (void)out_size; (void)ws_size;

  const float* x      = (const float*)d_in[0];
  const float* w7     = (const float*)d_in[1];
  const float* b7     = (const float*)d_in[2];
  const float* w3     = (const float*)d_in[3];
  const float* b3     = (const float*)d_in[4];
  // fusion param bases: l_fusion at 5, h_fusion at 14
  const float* fuse_w = (const float*)d_in[23];
  const float* fuse_b = (const float*)d_in[24];
  const float* cv_w   = (const float*)d_in[25];
  const float* cv_b   = (const float*)d_in[26];
  float* out = (float*)d_out;

  // bump allocator on d_ws
  char* ws = (char*)d_ws;
  size_t off = 0;
  auto alloc = [&](size_t bytes) -> void* {
    void* p = ws + off;
    off += (bytes + 255) & ~(size_t)255;
    return p;
  };

  _Float16* Dpad = (_Float16*)alloc(48 * 64 * 2);
  float*    mask = (float*)alloc(48 * 48 * 4);
  _Float16* W7t  = (_Float16*)alloc((size_t)49 * NC * NC * 2);
  _Float16* W3t  = (_Float16*)alloc((size_t)9 * NC * NC * 2);
  _Float16* Wxt[2], *Wzt[2], *Wot[2], *Wxpt[2], *Wct[2];
  for (int f = 0; f < 2; ++f) {
    Wxt[f]  = (_Float16*)alloc((size_t)NC * NC * 2);
    Wzt[f]  = (_Float16*)alloc((size_t)NC * NC * 2);
    Wot[f]  = (_Float16*)alloc((size_t)NC * NC * 2);
    Wxpt[f] = (_Float16*)alloc((size_t)32 * NC * 2);
    Wct[f]  = (_Float16*)alloc((size_t)32 * NC * 2);
  }
  _Float16* Wgt  = (_Float16*)alloc((size_t)NC * 192 * 2);
  _Float16* Wcvt = (_Float16*)alloc((size_t)NC * NC * 2);

  const size_t MxC = (size_t)NM * NC;
  _Float16* lfre = (_Float16*)alloc(MxC * 2);
  _Float16* hfre = (_Float16*)alloc(MxC * 2);
  _Float16* lspa = (_Float16*)alloc(MxC * 2);
  _Float16* hspa = (_Float16*)alloc(MxC * 2);
  float*    xf   = (float*)alloc(MxC * 4);
  _Float16* xh   = (_Float16*)alloc(MxC * 2);
  float*    zs   = (float*)alloc(MxC * 4);
  float*    dtB  = (float*)alloc((size_t)NM * (DTR + DST) * 4);
  float*    Cpb  = (float*)alloc((size_t)NM * DST * 4);
  float*    dtb  = (float*)alloc(MxC * 4);
  _Float16* yh   = (_Float16*)alloc(MxC * 2);
  _Float16* lfus = (_Float16*)alloc(MxC * 2);
  _Float16* hfus = (_Float16*)alloc(MxC * 2);
  _Float16* yfus = (_Float16*)alloc(MxC * 2);

  auto blk = [](int total) { return (total + 255) / 256; };

  // ---- prep ----
  prep_dct_kernel<<<blk(48 * 64 + 48 * 48), 256, 0, stream>>>(Dpad, mask);
  conv_w_prep_kernel<<<blk(49 * NC * NC), 256, 0, stream>>>(w7, W7t, 7);
  conv_w_prep_kernel<<<blk(9 * NC * NC), 256, 0, stream>>>(w3, W3t, 3);
  for (int f = 0; f < 2; ++f) {
    int base = 5 + 9 * f;
    const float* Wx  = (const float*)d_in[base + 0];
    const float* Wz  = (const float*)d_in[base + 1];
    const float* Wxp = (const float*)d_in[base + 2];
    const float* Wc  = (const float*)d_in[base + 5];
    const float* Wo  = (const float*)d_in[base + 8];
    mat_t_prep_kernel<<<blk(NC * NC), 256, 0, stream>>>(Wx, Wxt[f], NC, NC, NC);
    mat_t_prep_kernel<<<blk(NC * NC), 256, 0, stream>>>(Wz, Wzt[f], NC, NC, NC);
    mat_t_prep_kernel<<<blk(NC * NC), 256, 0, stream>>>(Wo, Wot[f], NC, NC, NC);
    mat_t_prep_kernel<<<blk(32 * NC), 256, 0, stream>>>(Wxp, Wxpt[f], NC, DTR + DST, 32);
    mat_t_prep_kernel<<<blk(32 * NC), 256, 0, stream>>>(Wc, Wct[f], NC, DST, 32);
  }
  mat_c_prep_kernel<<<blk(NC * 192), 256, 0, stream>>>(fuse_w, Wgt, NC * 192);
  mat_c_prep_kernel<<<blk(NC * NC), 256, 0, stream>>>(cv_w, Wcvt, NC * NC);

  // ---- DCT + masks -> lfre/hfre ----
  dct_kernel<<<NB * NC, 256, 0, stream>>>(x, Dpad, mask, lfre, hfre);

  // ---- spatial convs -> lspa/hspa ----
  conv_silu_kernel<7, 3><<<NM / 64, 256, 0, stream>>>(x, W7t, b7, lspa);
  conv_silu_kernel<3, 1><<<NM / 64, 256, 0, stream>>>(x, W3t, b3, hspa);

  // ---- two Mamba fusion paths ----
  const int G = NM / 64;  // 288 blocks
  for (int f = 0; f < 2; ++f) {
    int base = 5 + 9 * f;
    const float* Wdt   = (const float*)d_in[base + 3];
    const float* bdt   = (const float*)d_in[base + 4];
    const float* A_log = (const float*)d_in[base + 6];
    const float* Dp    = (const float*)d_in[base + 7];
    _Float16* spa = f ? hspa : lspa;
    _Float16* fre = f ? hfre : lfre;
    _Float16* fus = f ? hfus : lfus;

    // X = silu(U @ Wx) -> f32 + half
    gemm96_kernel<<<G, 256, 0, stream>>>(spa, nullptr, Wxt[f], nullptr,
                                         xf, xh, nullptr, nullptr, 0);
    // Zs = silu(V @ Wz) -> f32
    gemm96_kernel<<<G, 256, 0, stream>>>(fre, nullptr, Wzt[f], nullptr,
                                         zs, nullptr, nullptr, nullptr, 0);
    // dtB = X @ Wxp (N=22), Cp = V @ Wc (N=16)
    gemm32_kernel<<<G, 256, 0, stream>>>(xh, Wxpt[f], dtB, DTR + DST);
    gemm32_kernel<<<G, 256, 0, stream>>>(fre, Wct[f], Cpb, DST);
    // dt = softplus(dtB[:, :6] @ Wdt + bdt)
    dt_kernel<<<blk(NM * NC), 256, 0, stream>>>(dtB, Wdt, bdt, dtb);
    // selective scan (+ x*D, * silu(z)) -> yh (half)
    scan_kernel<<<NB, 96, 0, stream>>>(xf, dtb, dtB, Cpb, A_log, Dp, zs, yh);
    // fus = Y @ Wout -> half
    gemm96_kernel<<<G, 256, 0, stream>>>(yh, nullptr, Wot[f], nullptr,
                                         nullptr, fus, nullptr, nullptr, 1);
  }

  // ---- fuse gate: g = sigmoid(concat(l,h) @ Wg + b); y = g*l + (1-g)*h ----
  gemm96_kernel<<<G, 256, 0, stream>>>(lfus, hfus, Wgt, fuse_b,
                                       nullptr, yfus, lfus, hfus, 3);
  // ---- final 1x1 conv + silu -> d_out in [b][c][h][w] ----
  gemm96_kernel<<<G, 256, 0, stream>>>(yfus, nullptr, Wcvt, cv_b,
                                       out, nullptr, nullptr, nullptr, 2);
}
